// GCN_10694468567643
// MI455X (gfx1250) — matrix operand order repaired and checked by reading, not compiled, not run on hardware
//
#include <hip/hip_runtime.h>
#include <hip/hip_bf16.h>

// ---------------------------------------------------------------------------
// CDNA5 (gfx1250) GCN pipeline, bf16 WMMA GEMMs + fp32 atomic aggregation.
// All GEMM operands pre-converted to bf16 (padded so K % 32 == 0): inner loop
// is pure global_load_b128 + v_wmma_f32_16x16x32_bf16.
// ---------------------------------------------------------------------------

typedef __attribute__((ext_vector_type(16))) __bf16 v16bf;
typedef __attribute__((ext_vector_type(8)))  __bf16 bf16x8;
typedef __attribute__((ext_vector_type(8)))  float  v8f;

__device__ __forceinline__ float lrelu(float v) { return v > 0.f ? v : 0.1f * v; }

__device__ __forceinline__ bf16x8 zero8() {
  bf16x8 z;
#pragma unroll
  for (int i = 0; i < 8; ++i) z[i] = (__bf16)0.f;
  return z;
}

// A fragment: 16x32 (MxK) bf16, row-major source, row stride K (K%32==0).
// ISA 7.12.2 lane layout: lanes 0-15 / 16-31 share rows (M = lane&15) and split
// K: elems a[0..7] = K[kb + half*8 .. +7], a[8..15] = K[kb+16+half*8 .. +7].
template <bool GUARD>
__device__ __forceinline__ v16bf loadA(const __bf16* __restrict__ A, int m0, int kb,
                                       int K, int M, int lane) {
  const int half = lane >> 4;
  const int r    = m0 + (lane & 15);
  const bool valid = !GUARD || (r < M);
  const __bf16* row = A + (size_t)r * K;
  bf16x8 lo = valid ? *reinterpret_cast<const bf16x8*>(row + kb + (half << 3))      : zero8();
  bf16x8 hi = valid ? *reinterpret_cast<const bf16x8*>(row + kb + 16 + (half << 3)) : zero8();
  v16bf a;
#pragma unroll
  for (int i = 0; i < 8; ++i) { a[i] = lo[i]; a[i + 8] = hi[i]; }
  return a;
}

// B fragment: 32x16 (KxN), B[k][n] = W[n, k] (W row-major [FOUT, K]).
// Same K split as A: b[0..7] = K[kb + half*8 .. +7], b[8..15] = K[kb+16+half*8 .. +7].
__device__ __forceinline__ v16bf loadB(const __bf16* __restrict__ W, int n0, int kb,
                                       int K, int lane) {
  const int half = lane >> 4;
  const int n    = n0 + (lane & 15);
  const __bf16* row = W + (size_t)n * K + kb + (half << 3);
  bf16x8 lo = *reinterpret_cast<const bf16x8*>(row);
  bf16x8 hi = *reinterpret_cast<const bf16x8*>(row + 16);
  v16bf b;
#pragma unroll
  for (int i = 0; i < 8; ++i) { b[i] = lo[i]; b[i + 8] = hi[i]; }
  return b;
}

template <bool GUARD>
__device__ __forceinline__ void gemm_body(const __bf16* __restrict__ A,
                                          const __bf16* __restrict__ W,
                                          const float* __restrict__ bias, void* O,
                                          int M, int K, int FOUT, int m0, int n0,
                                          int lane, int act, int outBf16) {
  v8f acc0 = {0.f,0.f,0.f,0.f,0.f,0.f,0.f,0.f};
  v8f acc1 = acc0, acc2 = acc0, acc3 = acc0;

  for (int kb = 0; kb < K; kb += 32) {
    v16bf a  = loadA<GUARD>(A, m0, kb, K, M, lane);
    v16bf b0 = loadB(W, n0,      kb, K, lane);
    v16bf b1 = loadB(W, n0 + 16, kb, K, lane);
    v16bf b2 = loadB(W, n0 + 32, kb, K, lane);
    v16bf b3 = loadB(W, n0 + 48, kb, K, lane);
    acc0 = __builtin_amdgcn_wmma_f32_16x16x32_bf16(false, a, false, b0, (short)0, acc0, false, false);
    acc1 = __builtin_amdgcn_wmma_f32_16x16x32_bf16(false, a, false, b1, (short)0, acc1, false, false);
    acc2 = __builtin_amdgcn_wmma_f32_16x16x32_bf16(false, a, false, b2, (short)0, acc2, false, false);
    acc3 = __builtin_amdgcn_wmma_f32_16x16x32_bf16(false, a, false, b3, (short)0, acc3, false, false);
  }

  // C/D layout: VGPR r -> M = m0 + r + (lane>=16 ? 8 : 0), N = n0t + (lane&15).
  const int half = lane >> 4;
  const int ci   = lane & 15;
  v8f accs[4] = {acc0, acc1, acc2, acc3};
#pragma unroll
  for (int t = 0; t < 4; ++t) {
    const int col = n0 + (t << 4) + ci;
    const float bv = bias ? bias[col] : 0.f;
#pragma unroll
    for (int r = 0; r < 8; ++r) {
      const int row = m0 + r + (half << 3);
      if (!GUARD || row < M) {
        float v = accs[t][r] + bv;
        if (act) v = lrelu(v);
        const size_t idx = (size_t)row * FOUT + col;
        if (outBf16) ((__bf16*)O)[idx] = (__bf16)v;
        else         ((float*)O)[idx]  = v;
      }
    }
  }
}

// O[M, FOUT] = A[M, K] @ W[FOUT, K]^T (+bias, +leaky-relu). blockDim=256 (8 waves),
// each wave: 16 rows x 64 output features (4 WMMA accumulators). K % 32 == 0.
__global__ void wmma_gemm_kernel(const __bf16* __restrict__ A, const __bf16* __restrict__ W,
                                 const float* __restrict__ bias, void* O,
                                 int M, int K, int FOUT, int nGroups, int jobs,
                                 int act, int outBf16) {
  const int lane = threadIdx.x & 31;
  const int job  = blockIdx.x * 8 + (threadIdx.x >> 5);
  if (job >= jobs) return;
  const int mtile = job / nGroups;
  const int m0    = mtile << 4;
  const int n0    = (job - mtile * nGroups) << 6;
  if (m0 + 16 <= M)
    gemm_body<false>(A, W, bias, O, M, K, FOUT, m0, n0, lane, act, outBf16);
  else
    gemm_body<true>(A, W, bias, O, M, K, FOUT, m0, n0, lane, act, outBf16);
}

__global__ void fill0_kernel(float* __restrict__ p, long long n) {
  long long i = (long long)blockIdx.x * blockDim.x + threadIdx.x;
  if (i < n) p[i] = 0.f;
}

// out[r, c<Kin] = (bf16)in[r, c], zero-padded to Kout columns. n = rows*Kout.
__global__ void cvt_pad_kernel(const float* __restrict__ in, __bf16* __restrict__ out,
                               long long n, int Kin, int Kout) {
  long long i = (long long)blockIdx.x * blockDim.x + threadIdx.x;
  if (i >= n) return;
  long long r = i / Kout;
  int c = (int)(i - r * Kout);
  out[i] = (c < Kin) ? (__bf16)in[r * Kin + c] : (__bf16)0.f;
}

__global__ void cvt_kernel(const float* __restrict__ in, __bf16* __restrict__ out, long long n) {
  long long i = (long long)blockIdx.x * blockDim.x + threadIdx.x;
  if (i < n) out[i] = (__bf16)in[i];
}

__global__ void deg_kernel(const int* __restrict__ dst, float* __restrict__ deg, int E) {
  int t = blockIdx.x * blockDim.x + threadIdx.x;
  if (t < E) unsafeAtomicAdd(&deg[dst[t]], 1.0f);
}

__global__ void rsqrt_kernel(float* __restrict__ deg, int N) {
  int i = blockIdx.x * blockDim.x + threadIdx.x;
  if (i < N) deg[i] = rsqrtf(deg[i] + 1.0f);
}

// One thread per (edge, 4-feature chunk). F = 4 << cshift.
__global__ void agg_kernel(const float* __restrict__ h, const int* __restrict__ src,
                           const int* __restrict__ dst, const float* __restrict__ dis,
                           float* __restrict__ agg, int E, int cshift) {
  int t = blockIdx.x * blockDim.x + threadIdx.x;
  int e = t >> cshift;
  if (e >= E) return;
  int c = t & ((1 << cshift) - 1);
  int s = src[e], d = dst[e];
  float w = dis[s] * dis[d];
  const float4 v = *reinterpret_cast<const float4*>(h + (((size_t)s << cshift) + c) * 4);
  float* ap = agg + (((size_t)d << cshift) + c) * 4;
  unsafeAtomicAdd(ap + 0, v.x * w);
  unsafeAtomicAdd(ap + 1, v.y * w);
  unsafeAtomicAdd(ap + 2, v.z * w);
  unsafeAtomicAdd(ap + 3, v.w * w);
}

// act[i,f] = (bf16) lrelu(agg[i,f] + h[i,f]*dis[i]^2 + b[f]), F = 1 << fshift.
__global__ void self_bias_act_kernel(const float* __restrict__ agg, const float* __restrict__ h,
                                     const float* __restrict__ dis, const float* __restrict__ b,
                                     __bf16* __restrict__ act, int N, int fshift) {
  long long t = (long long)blockIdx.x * blockDim.x + threadIdx.x;
  if (t >= ((long long)N << fshift)) return;
  int i = (int)(t >> fshift);
  int f = (int)(t & ((1 << fshift) - 1));
  float dii = dis[i];
  float v = agg[t] + h[t] * dii * dii + b[f];
  act[t] = (__bf16)lrelu(v);
}

// out[g] = sigmoid(dot(g2[g,:], w) + b).  One wave32 per graph.
__global__ void final_kernel(const float* __restrict__ g2, const float* __restrict__ w,
                             const float* __restrict__ b, float* __restrict__ out,
                             int G, int F) {
  int g = blockIdx.x * (blockDim.x >> 5) + (threadIdx.x >> 5);
  int lane = threadIdx.x & 31;
  if (g >= G) return;
  const float* row = g2 + (size_t)g * F;
  float s = 0.f;
  for (int i = lane; i < F; i += 32) s += row[i] * w[i];
#pragma unroll
  for (int off = 16; off > 0; off >>= 1) s += __shfl_xor(s, off, 32);
  if (lane == 0) out[g] = 1.0f / (1.0f + __expf(-(s + b[0])));
}

extern "C" void kernel_launch(void* const* d_in, const int* in_sizes, int n_in,
                              void* d_out, int out_size, void* d_ws, size_t ws_size,
                              hipStream_t stream) {
  const float* x   = (const float*)d_in[0];
  const int*   ei  = (const int*)d_in[1];
  const float* W1  = (const float*)d_in[2];  const float* b1  = (const float*)d_in[3];
  const float* W2  = (const float*)d_in[4];  const float* b2  = (const float*)d_in[5];
  const float* W3  = (const float*)d_in[6];  const float* b3  = (const float*)d_in[7];
  const float* L1w = (const float*)d_in[8];  const float* L1b = (const float*)d_in[9];
  const float* L2w = (const float*)d_in[10]; const float* L2b = (const float*)d_in[11];
  const float* L3w = (const float*)d_in[12]; const float* L3b = (const float*)d_in[13];
  float* out = (float*)d_out;

  const int C  = 40;
  const int N  = in_sizes[0] / C;      // 200000
  const int E  = in_sizes[1] / 2;      // 1000000
  const int H1 = in_sizes[3];          // 128
  const int H2 = in_sizes[5];          // 256
  const int H3 = in_sizes[7];          // 128
  const int F1 = in_sizes[9];          // 512
  const int F2 = in_sizes[11];         // 128
  const int G  = N / 40;               // 5000
  const int KG = 40 * H3;              // 5120
  const int CP = 64;                   // C padded to K%32==0

  const int* src = ei;
  const int* dst = ei + E;

  // --- workspace carving (256B aligned) ---
  char* wp = (char*)d_ws;
  auto carve = [&](size_t bytes) { char* r = wp; wp += (bytes + 255) & ~(size_t)255; return r; };
  float*  deg   = (float*)carve((size_t)N * 4);
  float*  hbuf  = (float*)carve((size_t)N * H2 * 4);   // pre-agg h (fp32), reused per layer
  float*  aggb  = (float*)carve((size_t)N * H2 * 4);   // aggregation dest (fp32 atomics)
  __bf16* actA  = (__bf16*)carve((size_t)N * H2 * 2);  // act1, act3 (bf16)
  __bf16* actB  = (__bf16*)carve((size_t)N * H2 * 2);  // act2 (bf16)
  __bf16* xb    = (__bf16*)carve((size_t)N * CP * 2);  // x padded->bf16
  __bf16* W1b   = (__bf16*)carve((size_t)H1 * CP * 2);
  __bf16* W2b   = (__bf16*)carve((size_t)H2 * H1 * 2);
  __bf16* W3b   = (__bf16*)carve((size_t)H3 * H2 * 2);
  __bf16* L1wb  = (__bf16*)carve((size_t)F1 * KG * 2);
  __bf16* L2wb  = (__bf16*)carve((size_t)F2 * F1 * 2);
  __bf16* g1b   = (__bf16*)carve((size_t)G * F1 * 2);
  float*  g2    = (float*)carve((size_t)G * F2 * 4);

  auto nb = [](long long n, int bs) { return (unsigned)((n + bs - 1) / bs); };

  auto gemm = [&](const __bf16* A, const __bf16* W, const float* bias, void* O,
                  int M, int K, int FOUT, int act, int obf) {
    int nGroups = FOUT >> 6;
    int jobs    = ((M + 15) >> 4) * nGroups;
    wmma_gemm_kernel<<<nb(jobs, 8), 256, 0, stream>>>(A, W, bias, O, M, K, FOUT,
                                                      nGroups, jobs, act, obf);
  };

  // --- operand conversions (bf16; x/W1 zero-padded K=40 -> 64) ---
  cvt_pad_kernel<<<nb((long long)N * CP, 256), 256, 0, stream>>>(x, xb, (long long)N * CP, C, CP);
  cvt_pad_kernel<<<nb((long long)H1 * CP, 256), 256, 0, stream>>>(W1, W1b, (long long)H1 * CP, C, CP);
  cvt_kernel<<<nb((long long)H2 * H1, 256), 256, 0, stream>>>(W2, W2b, (long long)H2 * H1);
  cvt_kernel<<<nb((long long)H3 * H2, 256), 256, 0, stream>>>(W3, W3b, (long long)H3 * H2);
  cvt_kernel<<<nb((long long)F1 * KG, 256), 256, 0, stream>>>(L1w, L1wb, (long long)F1 * KG);
  cvt_kernel<<<nb((long long)F2 * F1, 256), 256, 0, stream>>>(L2w, L2wb, (long long)F2 * F1);

  // --- degrees: deg = rsqrt(segment_sum(1, dst) + 1) ---
  fill0_kernel<<<nb(N, 256), 256, 0, stream>>>(deg, N);
  deg_kernel<<<nb(E, 256), 256, 0, stream>>>(dst, deg, E);
  rsqrt_kernel<<<nb(N, 256), 256, 0, stream>>>(deg, N);

  auto conv = [&](const __bf16* actIn, int Kin, const __bf16* Wb, const float* bias,
                  int FOUT, __bf16* actOut) {
    gemm(actIn, Wb, nullptr, hbuf, N, Kin, FOUT, 0, 0);
    fill0_kernel<<<nb((long long)N * FOUT, 256), 256, 0, stream>>>(aggb, (long long)N * FOUT);
    int cs = __builtin_ctz(FOUT >> 2);
    agg_kernel<<<nb((long long)E << cs, 256), 256, 0, stream>>>(hbuf, src, dst, deg, aggb, E, cs);
    self_bias_act_kernel<<<nb((long long)N * FOUT, 256), 256, 0, stream>>>(
        aggb, hbuf, deg, bias, actOut, N, __builtin_ctz(FOUT));
  };

  conv(xb,   CP, W1b, b1, H1, actA);   // conv1: act1 -> actA
  conv(actA, H1, W2b, b2, H2, actB);   // conv2: act2 -> actB
  conv(actB, H2, W3b, b3, H3, actA);   // conv3: act3 -> actA  (act1 dead)

  // --- MLP head: act3 viewed as [G, 40*H3] ---
  gemm(actA, L1wb, L1b, g1b, G, KG, F1, 1, 1);   // bf16 out, fused bias+lrelu
  gemm(g1b,  L2wb, L2b, g2,  G, F1, F2, 1, 0);   // fp32 out, fused bias+lrelu
  final_kernel<<<nb((long long)G * 32, 256), 256, 0, stream>>>(g2, L3w, L3b, out, G, F2);
}
